// MultiHeadAttention_90245852823950
// MI455X (gfx1250) — matrix operand order repaired and checked by reading, not compiled, not run on hardware
//
#include <hip/hip_runtime.h>

// ---------------------------------------------------------------------------
// MultiHeadAttention forward for MI455X (gfx1250, wave32, WMMA bf16).
//
// Shapes: B=2, S=2048, D=768, H=12, DEPTH=64.
// Outputs (concat in d_out): out f32 [B,S,D] then attn f32 [B,H,S,S].
//
// Pipeline (all bf16 WMMA with f32 accumulate):
//   0) convert q,k,v and wq,wk,wv,wo from f32 to bf16 (packed in ws)
//   1) Qp = q @ wq^T + b   -> bf16 [B,H,S,64]
//      Kp = k @ wk^T + b   -> bf16 [B,H,S,64]
//      Vt = v @ wv^T + b   -> bf16 [B,H,64,S]   (transposed for PV B-operand)
//   2) attention per (b,h,16-row q tile), 4 waves/block sharing K/V tiles
//      staged through LDS with double-buffered GLOBAL_LOAD_ASYNC_TO_LDS_B128:
//        logits = Q K^T / 8  -> attn buffer (f32, also the required output)
//        masked softmax in place (3 streaming passes)
//        ctx = P V           -> bf16 [B,S,768]
//   3) out = ctx @ wo^T + b -> f32
// ---------------------------------------------------------------------------

#define BB   2
#define SS   2048
#define DD   768
#define HH   12
#define DEP  64
#define MROW (BB * SS)          // 4096 rows for the projection GEMMs
#define NTIL (DD / 16)          // 48
#define MTIL (MROW / 16)        // 256

typedef __attribute__((ext_vector_type(16))) __bf16 v16bf;
typedef __attribute__((ext_vector_type(8)))  float  v8f;

union V16 {
    v16bf v;
    uint4 q[2];
    unsigned short s[16];
};

__device__ __forceinline__ unsigned short f32_to_bf16(float f) {
    union { float f; unsigned int u; } x; x.f = f;
    unsigned int lsb = (x.u >> 16) & 1u;
    x.u += 0x7fffu + lsb;                  // round to nearest even
    return (unsigned short)(x.u >> 16);
}

__device__ __forceinline__ void pack4(unsigned short* d, float4 f) {
    d[0] = f32_to_bf16(f.x); d[1] = f32_to_bf16(f.y);
    d[2] = f32_to_bf16(f.z); d[3] = f32_to_bf16(f.w);
}

// gfx1250 async copy: 16 bytes global -> LDS per lane, tracked by ASYNCcnt.
__device__ __forceinline__ void async_copy16(unsigned lds_off, const void* gsrc) {
    asm volatile("global_load_async_to_lds_b128 %0, %1, off"
                 :: "v"(lds_off), "v"(gsrc) : "memory");
}
__device__ __forceinline__ void wait_asynccnt0() {
    asm volatile("s_wait_asynccnt 0" ::: "memory");
}
__device__ __forceinline__ void wait_storecnt0() {
    asm volatile("s_wait_storecnt 0" ::: "memory");
}

// ---------------------------------------------------------------------------
// Kernel 0: elementwise f32 -> bf16 conversion (grid-stride)
// ---------------------------------------------------------------------------
__global__ void cvt_bf16_kernel(const float* __restrict__ src,
                                unsigned short* __restrict__ dst, int n) {
    for (int i = blockIdx.x * blockDim.x + threadIdx.x; i < n;
         i += gridDim.x * blockDim.x)
        dst[i] = f32_to_bf16(src[i]);
}

// ---------------------------------------------------------------------------
// Generic GEMM: C[m,n] = sum_k A[m,k] * W[n,k] + bias[n]
//   A bf16 [4096,768], W bf16 [768,768] (row-major, i.e. W^T GEMM), bias f32.
// One wave -> one 16x16 tile, K-loop of 24 x v_wmma_f32_16x16x32_bf16.
// MODE 0: dst bf16 head-major [B,H,S,64]   (Q, K projections)
// MODE 1: dst bf16 head-major transposed [B,H,64,S]  (V projection)
// MODE 2: dst f32 [4096,768]               (output projection)
// ---------------------------------------------------------------------------
template <int MODE>
__global__ __launch_bounds__(128)
void gemm16_kernel(const unsigned short* __restrict__ A,
                   const unsigned short* __restrict__ W,
                   const float* __restrict__ bias,
                   void* __restrict__ dstv) {
    const int lane = threadIdx.x & 31;
    const int tile = blockIdx.x * 4 + (threadIdx.x >> 5);   // 12288 tiles
    const int tm = tile / NTIL;
    const int tn = tile % NTIL;
    const int half = lane >> 4;
    const int l16  = lane & 15;

    // A: lane = M row (rows 0-15 duplicated across halves; halves carry
    //    different K sub-ranges: {8h..8h+7} U {16+8h..16+8h+7} per 32-K step).
    // B: lane = N col of output = row of W; same K sub-ranges as A.
    const unsigned short* arow = A + (size_t)(tm * 16 + l16) * DD;
    const unsigned short* wrow = W + (size_t)(tn * 16 + l16) * DD;

    v8f c = {};
#pragma unroll 4
    for (int k0 = 0; k0 < DD; k0 += 32) {
        V16 a, b;
        a.q[0] = *(const uint4*)(arow + k0 + 8 * half);
        a.q[1] = *(const uint4*)(arow + k0 + 16 + 8 * half);
        b.q[0] = *(const uint4*)(wrow + k0 + 8 * half);
        b.q[1] = *(const uint4*)(wrow + k0 + 16 + 8 * half);
        __builtin_prefetch(arow + k0 + 256, 0, 1);   // global_prefetch_b8
        __builtin_prefetch(wrow + k0 + 256, 0, 1);
        c = __builtin_amdgcn_wmma_f32_16x16x32_bf16(
                false, a.v, false, b.v, (short)0, c, false, false);
    }

    // C layout: lane = N col; VGPR r holds row (8*half + r).
    const int ng = tn * 16 + l16;
    const float bv = bias[ng];
#pragma unroll
    for (int r = 0; r < 8; ++r) {
        const int mg = tm * 16 + 8 * half + r;
        const float val = c[r] + bv;
        if (MODE == 2) {
            ((float*)dstv)[(size_t)mg * DD + ng] = val;
        } else {
            const int bb = mg >> 11, ss = mg & (SS - 1);
            const int hh = ng >> 6,  dd = ng & (DEP - 1);
            unsigned short* dst = (unsigned short*)dstv;
            const size_t idx = (MODE == 0)
                ? ((((size_t)bb * HH + hh) * SS + ss) * DEP + dd)
                : ((((size_t)bb * HH + hh) * DEP + dd) * SS + ss);
            dst[idx] = f32_to_bf16(val);
        }
    }
}

// ---------------------------------------------------------------------------
// Attention: one wave per (b, h, 16-row q tile); 4 waves/block cover 4
// consecutive q tiles of the SAME (b,h), so K/V tiles are staged once per
// block in LDS via double-buffered async copies (ASYNCcnt) instead of being
// re-fetched by every wave.
// ---------------------------------------------------------------------------
__global__ __launch_bounds__(128)
void attn_kernel(const unsigned short* __restrict__ Qp,
                 const unsigned short* __restrict__ Kp,
                 const unsigned short* __restrict__ Vt,
                 const int* __restrict__ mask,
                 float* __restrict__ attn,
                 unsigned short* __restrict__ ctx) {
    __shared__ unsigned short lds[4096];        // 8 KB, double-buffered tiles
    const unsigned lds_base = (unsigned)(uintptr_t)lds;  // LDS aperture: low
                                                         // 32 bits = offset
    const int tid  = threadIdx.x;
    const int lane = tid & 31;
    const int wid  = blockIdx.x * 4 + (tid >> 5);
    const int qt = wid & 127;
    const int h  = (wid >> 7) % HH;
    const int b  = wid / (128 * HH);
    const int half = lane >> 4;
    const int l16  = lane & 15;

    const unsigned short* Qh = Qp + ((size_t)b * HH + h) * SS * DEP;
    const unsigned short* Kh = Kp + ((size_t)b * HH + h) * SS * DEP;
    const unsigned short* Vh = Vt + ((size_t)b * HH + h) * DEP * SS;
    float* arow = attn + (((size_t)b * HH + h) * SS + (size_t)qt * 16) * SS;
    const int* mp = mask + (size_t)b * SS;

    // ---- Phase A: logits = (Q K^T) * (1/sqrt(64)) -> attn buffer -----------
    V16 a0, a1;
    {
        const unsigned short* qrow = Qh + (size_t)(qt * 16 + l16) * DEP;
        a0.q[0] = *(const uint4*)(qrow + 8 * half);
        a0.q[1] = *(const uint4*)(qrow + 16 + 8 * half);
        a1.q[0] = *(const uint4*)(qrow + 32 + 8 * half);
        a1.q[1] = *(const uint4*)(qrow + 48 + 8 * half);
    }

    // K tile nt = rows [nt*16, nt*16+16) x 64 bf16 = 2 KB contiguous block.
    // 128 threads x 16B async = whole tile in one shot; double buffer.
    {
        // prefetch tile 0 into buffer 0
        async_copy16(lds_base + tid * 16, Kh + tid * 8);
        int buf = 0;
        for (int nt = 0; nt < SS / 16; ++nt) {
            wait_asynccnt0();          // own lanes' share of current fill
            __syncthreads();           // whole buffer visible block-wide
            if (nt + 1 < SS / 16)
                async_copy16(lds_base + (buf ^ 1) * 2048 + tid * 16,
                             Kh + (size_t)(nt + 1) * 1024 + tid * 8);
            const unsigned short* kt = lds + buf * 1024;   // [16][64]
            V16 b0, b1;
            b0.q[0] = *(const uint4*)(kt + l16 * 64 + 8 * half);
            b0.q[1] = *(const uint4*)(kt + l16 * 64 + 16 + 8 * half);
            b1.q[0] = *(const uint4*)(kt + l16 * 64 + 32 + 8 * half);
            b1.q[1] = *(const uint4*)(kt + l16 * 64 + 48 + 8 * half);
            v8f c = {};
            c = __builtin_amdgcn_wmma_f32_16x16x32_bf16(
                    false, a0.v, false, b0.v, (short)0, c, false, false);
            c = __builtin_amdgcn_wmma_f32_16x16x32_bf16(
                    false, a1.v, false, b1.v, (short)0, c, false, false);
#pragma unroll
            for (int r = 0; r < 8; ++r)
                arow[(size_t)(8 * half + r) * SS + nt * 16 + l16] =
                    c[r] * 0.125f;
            buf ^= 1;
        }
    }
    wait_storecnt0();

    // ---- Phase B: masked softmax, in place, per row ------------------------
    // lane owns (row = l16, column half = half*1024..+1023); cross-half
    // reduction via shfl_xor(16) on wave32.
    {
        float* pr = arow + (size_t)l16 * SS;
        const int cbase = half * 1024;

        float mx = -1e30f;
        for (int c4 = 0; c4 < 256; ++c4) {
            const int col = cbase + (c4 << 2);
            const float4 x = *(const float4*)(pr + col);
            const int4  mk = *(const int4*)(mp + col);
            mx = fmaxf(mx, mk.x ? x.x : -1e30f);
            mx = fmaxf(mx, mk.y ? x.y : -1e30f);
            mx = fmaxf(mx, mk.z ? x.z : -1e30f);
            mx = fmaxf(mx, mk.w ? x.w : -1e30f);
        }
        mx = fmaxf(mx, __shfl_xor(mx, 16, 32));

        float sum = 0.f;
        for (int c4 = 0; c4 < 256; ++c4) {
            const int col = cbase + (c4 << 2);
            const float4 x = *(const float4*)(pr + col);
            const int4  mk = *(const int4*)(mp + col);
            float4 p;
            p.x = mk.x ? __expf(x.x - mx) : 0.f;
            p.y = mk.y ? __expf(x.y - mx) : 0.f;
            p.z = mk.z ? __expf(x.z - mx) : 0.f;
            p.w = mk.w ? __expf(x.w - mx) : 0.f;
            sum += p.x + p.y + p.z + p.w;
            *(float4*)(pr + col) = p;
        }
        sum += __shfl_xor(sum, 16, 32);
        const float inv = sum > 0.f ? 1.f / sum : 0.f;

        wait_storecnt0();
        for (int c4 = 0; c4 < 256; ++c4) {
            const int col = cbase + (c4 << 2);
            float4 x = *(const float4*)(pr + col);
            x.x *= inv; x.y *= inv; x.z *= inv; x.w *= inv;
            *(float4*)(pr + col) = x;
        }
    }
    wait_storecnt0();
    __syncthreads();   // everyone done with Phase A's LDS before V staging

    // ---- Phase C: ctx = P @ V  (P f32 from attn buffer -> bf16 A operand) --
    // V chunk per 32-K step: 64 rows (all d) x 32 k bf16 = 4 KB; 128 threads
    // stage 2 x 16B each; double buffer.  LDS layout: vbuf[row][k'] row=d idx,
    // k' = 0..31, row stride 32 elements.
    v8f acc[4] = {{}, {}, {}, {}};
    const float* prA = arow + (size_t)l16 * SS;
    {
        // prefetch k0 = 0 into buffer 0
        {
            const int c0 = tid, c1 = tid + 128;
            async_copy16(lds_base + c0 * 16,
                         Vh + (size_t)(c0 >> 2) * SS + (c0 & 3) * 8);
            async_copy16(lds_base + c1 * 16,
                         Vh + (size_t)(c1 >> 2) * SS + (c1 & 3) * 8);
        }
        int buf = 0;
        for (int k0 = 0; k0 < SS; k0 += 32) {
            wait_asynccnt0();
            __syncthreads();
            if (k0 + 32 < SS) {
                const int c0 = tid, c1 = tid + 128;
                async_copy16(lds_base + (buf ^ 1) * 4096 + c0 * 16,
                             Vh + (size_t)(c0 >> 2) * SS + k0 + 32 + (c0 & 3) * 8);
                async_copy16(lds_base + (buf ^ 1) * 4096 + c1 * 16,
                             Vh + (size_t)(c1 >> 2) * SS + k0 + 32 + (c1 & 3) * 8);
            }

            // A: probs row l16, K chunk per A layout, f32 -> bf16
            const float4 f0 = *(const float4*)(prA + k0 + 8 * half);
            const float4 f1 = *(const float4*)(prA + k0 + 8 * half + 4);
            const float4 f2 = *(const float4*)(prA + k0 + 16 + 8 * half);
            const float4 f3 = *(const float4*)(prA + k0 + 16 + 8 * half + 4);
            V16 a;
            pack4(a.s + 0,  f0);
            pack4(a.s + 4,  f1);
            pack4(a.s + 8,  f2);
            pack4(a.s + 12, f3);

            const unsigned short* vt = lds + buf * 2048;   // [64][32]
#pragma unroll
            for (int dt = 0; dt < 4; ++dt) {
                V16 bb;
                bb.q[0] = *(const uint4*)(vt + (dt * 16 + l16) * 32 + 8 * half);
                bb.q[1] = *(const uint4*)(vt + (dt * 16 + l16) * 32 + 16 + 8 * half);
                acc[dt] = __builtin_amdgcn_wmma_f32_16x16x32_bf16(
                              false, a.v, false, bb.v, (short)0, acc[dt],
                              false, false);
            }
            buf ^= 1;
        }
    }

    // store ctx bf16 at [b, s, h*64 + d]
#pragma unroll
    for (int dt = 0; dt < 4; ++dt) {
#pragma unroll
        for (int r = 0; r < 8; ++r) {
            const int s = qt * 16 + 8 * half + r;
            const int d = h * DEP + dt * 16 + l16;
            ctx[((size_t)b * SS + s) * DD + d] = f32_to_bf16(acc[dt][r]);
        }
    }
}

// ---------------------------------------------------------------------------
// Host-side launch
// ---------------------------------------------------------------------------
extern "C" void kernel_launch(void* const* d_in, const int* in_sizes, int n_in,
                              void* d_out, int out_size, void* d_ws, size_t ws_size,
                              hipStream_t stream) {
    (void)in_sizes; (void)n_in; (void)out_size; (void)ws_size;

    const float* q    = (const float*)d_in[0];
    const float* k    = (const float*)d_in[1];
    const float* v    = (const float*)d_in[2];
    const int*   mask = (const int*)d_in[3];
    const float* wq_w = (const float*)d_in[4];
    const float* wq_b = (const float*)d_in[5];
    const float* wk_w = (const float*)d_in[6];
    const float* wk_b = (const float*)d_in[7];
    const float* wv_w = (const float*)d_in[8];
    const float* wv_b = (const float*)d_in[9];
    const float* wo_w = (const float*)d_in[10];
    const float* wo_b = (const float*)d_in[11];

    float* out_f32  = (float*)d_out;                    // [B,S,D]
    float* attn_out = out_f32 + (size_t)BB * SS * DD;   // [B,H,S,S]

    // workspace carve (bf16 stored as unsigned short)
    const size_t NA = (size_t)BB * SS * DD;   // 3,145,728
    const size_t NW = (size_t)DD * DD;        //   589,824
    unsigned short* p   = (unsigned short*)d_ws;
    unsigned short* qb  = p;  p += NA;
    unsigned short* kb  = p;  p += NA;
    unsigned short* vb  = p;  p += NA;
    unsigned short* wqb = p;  p += NW;
    unsigned short* wkb = p;  p += NW;
    unsigned short* wvb = p;  p += NW;
    unsigned short* wob = p;  p += NW;
    unsigned short* Qp  = p;  p += NA;        // [B,H,S,64]
    unsigned short* Kp  = p;  p += NA;        // [B,H,S,64]
    unsigned short* Vt  = p;  p += NA;        // [B,H,64,S]
    unsigned short* ctx = p;  p += NA;        // [B,S,768]

    // 0) f32 -> bf16 packing
    cvt_bf16_kernel<<<2048, 256, 0, stream>>>(q,    qb,  (int)NA);
    cvt_bf16_kernel<<<2048, 256, 0, stream>>>(k,    kb,  (int)NA);
    cvt_bf16_kernel<<<2048, 256, 0, stream>>>(v,    vb,  (int)NA);
    cvt_bf16_kernel<<<1024, 256, 0, stream>>>(wq_w, wqb, (int)NW);
    cvt_bf16_kernel<<<1024, 256, 0, stream>>>(wk_w, wkb, (int)NW);
    cvt_bf16_kernel<<<1024, 256, 0, stream>>>(wv_w, wvb, (int)NW);
    cvt_bf16_kernel<<<1024, 256, 0, stream>>>(wo_w, wob, (int)NW);

    // 1) projections (12288 tiles / 4 waves per block)
    const int gemm_blocks = (MTIL * NTIL) / 4;   // 3072
    gemm16_kernel<0><<<gemm_blocks, 128, 0, stream>>>(qb, wqb, wq_b, Qp);
    gemm16_kernel<0><<<gemm_blocks, 128, 0, stream>>>(kb, wkb, wk_b, Kp);
    gemm16_kernel<1><<<gemm_blocks, 128, 0, stream>>>(vb, wvb, wv_b, Vt);

    // 2) attention (3072 waves / 4 per block, block shares K/V via LDS)
    attn_kernel<<<768, 128, 0, stream>>>(Qp, Kp, Vt, mask, attn_out, ctx);

    // 3) output projection -> f32
    gemm16_kernel<2><<<gemm_blocks, 128, 0, stream>>>(ctx, wob, wo_b, (void*)out_f32);
}
